// EfficientRelativePositionalEncoding_79061757985079
// MI455X (gfx1250) — matrix operand order repaired and checked by reading, not compiled, not run on hardware
//
#include <hip/hip_runtime.h>

typedef float    v2f  __attribute__((ext_vector_type(2)));
typedef float    v4f  __attribute__((ext_vector_type(4)));
typedef float    v8f  __attribute__((ext_vector_type(8)));
typedef _Float16 v16h __attribute__((ext_vector_type(16)));

#define N_PTS 2048
#define BATCH 2
#define HEADS 8

// ---------------------------------------------------------------------------
// Workspace layout (float units unless noted):
//   ws[  0..127] : stage-1 B operands  ws_b1[tile(2)][vgpr(2)][lane(32)]  (f32)
//   ws[128..191] : stage-1 C bias splat ws_c1[tile(2)][lane(32)]          (f32)
//   ws[192..199] : fused bias b2cat[8]                                     (f32)
//   ws[256..511] : (as uint) stage-2 B operand, [lane(32)][vgpr(8)] packed f16x2
// ---------------------------------------------------------------------------

__global__ void erp_prep_kernel(const float* Wd1, const float* bd1,
                                const float* Wd2, const float* bd2,
                                const float* Wr1, const float* br1,
                                const float* Wr2, const float* br2,
                                const float* Wf,  const float* bf,
                                float* ws) {
  if (threadIdx.x != 0 || blockIdx.x != 0) return;

  // Block-diagonal first-layer weights over features u = [dist, ux, uy, uz]
  float W1[4][32];
  float b1[32];
  for (int h = 0; h < 32; ++h) {
    if (h < 16) {
      W1[0][h] = Wd1[h]; W1[1][h] = 0.f; W1[2][h] = 0.f; W1[3][h] = 0.f;
      b1[h] = bd1[h];
    } else {
      int c = h - 16;
      W1[0][h] = 0.f;
      W1[1][h] = Wr1[0 * 16 + c];
      W1[2][h] = Wr1[1 * 16 + c];
      W1[3][h] = Wr1[2 * 16 + c];
      b1[h] = br1[c];
    }
  }

  // Fold fusion layer: W2cat[32][8], b2cat[8]
  float W2[32][8];
  float b2[8];
  for (int o = 0; o < 8; ++o) {
    float acc = bf[o];
    for (int m = 0; m < 8; ++m)
      acc += bd2[m] * Wf[m * 8 + o] + br2[m] * Wf[(8 + m) * 8 + o];
    b2[o] = acc;
  }
  for (int h = 0; h < 16; ++h) {
    for (int o = 0; o < 8; ++o) {
      float ad = 0.f, ar = 0.f;
      for (int m = 0; m < 8; ++m) {
        ad += Wd2[h * 8 + m] * Wf[m * 8 + o];
        ar += Wr2[h * 8 + m] * Wf[(8 + m) * 8 + o];
      }
      W2[h][o]      = ad;
      W2[16 + h][o] = ar;
    }
  }

  // Stage-1 B tiles (f32 4x16 each): lane l -> column n = l&15 (hidden unit),
  // vgpr v -> K = v + 2*(l>=16)
  for (int t = 0; t < 2; ++t)
    for (int v = 0; v < 2; ++v)
      for (int l = 0; l < 32; ++l) {
        int hdn = t * 16 + (l & 15);
        int K   = v + 2 * (l >> 4);
        ws[t * 64 + v * 32 + l] = W1[K][hdn];
      }

  // Stage-1 C bias splat per lane
  for (int t = 0; t < 2; ++t)
    for (int l = 0; l < 32; ++l)
      ws[128 + t * 32 + l] = b1[t * 16 + (l & 15)];

  for (int o = 0; o < 8; ++o) ws[192 + o] = b2[o];

  // Stage-2 B operand (f16 32x16, cols 8..15 zero-padded):
  // lane l -> column n = l&15, K half = 16*(l>=16); vgpr v -> K = base+2v,2v+1
  unsigned* wb2 = (unsigned*)(ws + 256);
  for (int l = 0; l < 32; ++l) {
    int n  = l & 15;
    int kb = (l >> 4) * 16;
    for (int v = 0; v < 8; ++v) {
      float f0 = (n < 8) ? W2[kb + 2 * v][n]     : 0.f;
      float f1 = (n < 8) ? W2[kb + 2 * v + 1][n] : 0.f;
      union { _Float16 h[2]; unsigned u; } pk;
      pk.h[0] = (_Float16)f0;
      pk.h[1] = (_Float16)f1;
      wb2[l * 8 + v] = pk.u;
    }
  }
}

// ---------------------------------------------------------------------------
// Main kernel: one wave per (b, i) row, 16 j's per WMMA tile, 128 tiles/wave.
// xyz load for tile jt+1 is software-pipelined over tile jt's compute chain;
// output is streamed with non-temporal stores (write-once, never re-read).
// ---------------------------------------------------------------------------
__global__ __launch_bounds__(128) void erp_main_kernel(
    const float* __restrict__ xyz, const float* __restrict__ ws,
    float* __restrict__ out) {
  __shared__ __align__(16) float lds[4][16][34];  // [wave][pair][hidden] + pad

  const int lane = threadIdx.x & 31;
  const int wv   = threadIdx.x >> 5;
  const int row  = blockIdx.x * 4 + wv;   // 0 .. B*N-1
  const int b    = row >> 11;             // N = 2048
  const int i    = row & (N_PTS - 1);
  const int p    = lane & 15;             // pair-in-tile / column index
  const int hi   = lane >> 4;             // lane half

  // Constant operands (L2-resident after first wave)
  v2f b1a, b1b;
  b1a.x = ws[0 * 64 + 0 * 32 + lane];
  b1a.y = ws[0 * 64 + 1 * 32 + lane];
  b1b.x = ws[1 * 64 + 0 * 32 + lane];
  b1b.y = ws[1 * 64 + 1 * 32 + lane];
  const float c1a_s = ws[128 + lane];
  const float c1b_s = ws[128 + 32 + lane];
  const float c2_s  = (p < 8) ? ws[192 + p] : 0.f;

  v16h b2m;
  {
    const uint4* pb = (const uint4*)((const unsigned*)(ws + 256) + lane * 8);
    uint4 q0 = pb[0], q1 = pb[1];
    uint4* d = (uint4*)&b2m;
    d[0] = q0; d[1] = q1;
  }

  const float xi0 = xyz[(b * N_PTS + i) * 3 + 0];
  const float xi1 = xyz[(b * N_PTS + i) * 3 + 1];
  const float xi2 = xyz[(b * N_PTS + i) * 3 + 2];

  const float* xrow = xyz + (size_t)b * N_PTS * 3;

  // Software pipeline: preload tile 0's coordinates.
  float cx = xrow[(0 + p) * 3 + 0];
  float cy = xrow[(0 + p) * 3 + 1];
  float cz = xrow[(0 + p) * 3 + 2];

  for (int jt = 0; jt < N_PTS / 16; ++jt) {
    // Issue next tile's load now; waited on ~an entire tile of compute later.
    const int jn = ((jt + 1) & (N_PTS / 16 - 1)) * 16 + p;
    const float* xn = xrow + jn * 3;
    const float nx = xn[0];
    const float ny = xn[1];
    const float nz = xn[2];

    const float dx = cx - xi0;
    const float dy = cy - xi1;
    const float dz = cz - xi2;
    // fast transcendental path: v_sqrt_f32 + v_rcp_f32 (stage-2 is f16 anyway)
    const float d   = __builtin_amdgcn_sqrtf(dx * dx + dy * dy + dz * dz);
    const float inv = __builtin_amdgcn_rcpf(d + 1e-7f);
    const float ux = dx * inv, uy = dy * inv, uz = dz * inv;

    // A (16x4 f32): lanes 0-15 carry K={0,1}=(dist,ux); lanes 16-31 K={2,3}=(uy,uz)
    v2f a1;
    a1.x = hi ? uy : d;
    a1.y = hi ? uz : ux;

    v8f c1a, c1b;
#pragma unroll
    for (int v = 0; v < 8; ++v) { c1a[v] = c1a_s; c1b[v] = c1b_s; }

    c1a = __builtin_amdgcn_wmma_f32_16x16x4_f32(false, a1, false, b1a,
                                                (short)0, c1a, false, false);
    c1b = __builtin_amdgcn_wmma_f32_16x16x4_f32(false, a1, false, b1b,
                                                (short)0, c1b, false, false);

    // ReLU + transpose through LDS: D layout (pair across vgprs, hidden across
    // lanes) -> A layout (pair across lanes, hidden across vgprs as f16)
#pragma unroll
    for (int v = 0; v < 8; ++v) {
      lds[wv][v + 8 * hi][p]      = fmaxf(c1a[v], 0.f);
      lds[wv][v + 8 * hi][16 + p] = fmaxf(c1b[v], 0.f);
    }
    asm volatile("s_wait_dscnt 0" ::: "memory");

    v16h a2;
#pragma unroll
    for (int v = 0; v < 8; ++v) {
      const int kb = 2 * v + 16 * hi;
      const float2 f = *(const float2*)&lds[wv][p][kb];
      a2[2 * v]     = (_Float16)f.x;
      a2[2 * v + 1] = (_Float16)f.y;
    }

    v8f c2;
#pragma unroll
    for (int v = 0; v < 8; ++v) c2[v] = c2_s;

    c2 = __builtin_amdgcn_wmma_f32_16x16x32_f16(false, a2, false, b2m,
                                                (short)0, c2, false, false);

    // D: lane n<8 holds head H=n for 8 consecutive j (j = jt*16 + 8*hi + v)
    if (p < 8) {
      float* o = out + (((size_t)(b * HEADS + p) * N_PTS + i) * N_PTS) +
                 (size_t)jt * 16 + 8 * hi;
      const v4f* oc = (const v4f*)&c2;
      __builtin_nontemporal_store(oc[0], (v4f*)(o));
      __builtin_nontemporal_store(oc[1], (v4f*)(o + 4));
    }

    cx = nx; cy = ny; cz = nz;
  }
}

extern "C" void kernel_launch(void* const* d_in, const int* in_sizes, int n_in,
                              void* d_out, int out_size, void* d_ws,
                              size_t ws_size, hipStream_t stream) {
  (void)in_sizes; (void)n_in; (void)out_size; (void)ws_size;
  const float* xyz = (const float*)d_in[0];
  const float* Wd1 = (const float*)d_in[1];
  const float* bd1 = (const float*)d_in[2];
  const float* Wd2 = (const float*)d_in[3];
  const float* bd2 = (const float*)d_in[4];
  const float* Wr1 = (const float*)d_in[5];
  const float* br1 = (const float*)d_in[6];
  const float* Wr2 = (const float*)d_in[7];
  const float* br2 = (const float*)d_in[8];
  const float* Wf  = (const float*)d_in[9];
  const float* bf  = (const float*)d_in[10];
  float* ws = (float*)d_ws;

  erp_prep_kernel<<<1, 1, 0, stream>>>(Wd1, bd1, Wd2, bd2, Wr1, br1, Wr2, br2,
                                       Wf, bf, ws);
  // one wave per (b,i): B*N/4 blocks of 4 waves
  erp_main_kernel<<<(BATCH * N_PTS) / 4, 128, 0, stream>>>(xyz, ws,
                                                           (float*)d_out);
}
